// OctFormerBlock_68204080660879
// MI455X (gfx1250) — compile-verified, matrix operand order checked by reading, not run on hardware
//
#include <hip/hip_runtime.h>
#include <hip/hip_bf16.h>
#include <math.h>

// ---------------- problem constants ----------------
#define N_WIN 8192
#define KW    32
#define C_DIM 192
#define H_N   6
#define HDIM  32
#define HID   768
#define MTOT  (N_WIN * KW)          // 262144 tokens
#define POSB  25
#define RPEN  51                    // 2*POSB+1
#define ATT_SCALE 0.17677669529663689f   // 32^-0.5

typedef __attribute__((ext_vector_type(16))) _Float16 v16h;
typedef __attribute__((ext_vector_type(8)))  _Float16 v8h;
typedef __attribute__((ext_vector_type(8)))  float    v8f;
typedef __attribute__((ext_vector_type(4)))  int      v4i;

// -------- CDNA5 async global->LDS copy (ASYNCcnt path), guarded fallback --------
#if __has_builtin(__builtin_amdgcn_global_load_async_to_lds_b128) && \
    __has_builtin(__builtin_amdgcn_s_wait_asynccnt)
#define HAVE_ASYNC 1
typedef __attribute__((address_space(1))) v4i* as1_v4i;
typedef __attribute__((address_space(3))) v4i* as3_v4i;
#else
#define HAVE_ASYNC 0
#endif

__device__ inline void cp16_lds(_Float16* dst, const _Float16* src) {
#if HAVE_ASYNC
  __builtin_amdgcn_global_load_async_to_lds_b128((as1_v4i)src, (as3_v4i)dst, 0, 0);
#else
  *(v8h*)dst = *(const v8h*)src;
#endif
}

__device__ inline void async_tile_wait() {
#if HAVE_ASYNC
  __builtin_amdgcn_s_wait_asynccnt(0);
#endif
}

// Load a 16x32 f16 WMMA operand fragment from LDS.
// Tile is row-major with row stride `ld` halves; rows are the 16 M (or N) indices.
// Per ISA: lanes 0-15 -> K {0..7, 16..23}; lanes 16-31 -> K {8..15, 24..31}.
__device__ inline v16h load_frag(const _Float16* base, int ld, int lane) {
  const int r  = lane & 15;
  const int kb = (lane >> 4) << 3;            // 0 or 8
  v8h lo = *(const v8h*)(base + r * ld + kb);
  v8h hi = *(const v8h*)(base + r * ld + kb + 16);
  v16h f;
#pragma unroll
  for (int i = 0; i < 8; ++i) { f[i] = lo[i]; f[i + 8] = hi[i]; }
  return f;
}

// ---------------- weight convert: fp32 [Kc][Nc] -> f16 transposed [Nc][Kc] ----------------
__global__ __launch_bounds__(256) void convert_w(const float* __restrict__ w,
                                                 _Float16* __restrict__ wt,
                                                 int Kc, int Nc) {
  int idx = blockIdx.x * 256 + threadIdx.x;
  if (idx < Kc * Nc) {
    int n = idx / Kc, k = idx - n * Kc;
    wt[idx] = (_Float16)w[k * Nc + n];
  }
}

// ---------------- layernorm: wave-per-token, f32 in -> f16 out ----------------
__global__ __launch_bounds__(256) void ln_kernel(const float* __restrict__ x,
                                                 const float* __restrict__ g,
                                                 const float* __restrict__ b,
                                                 _Float16* __restrict__ out) {
  const int token = blockIdx.x * 8 + (threadIdx.x >> 5);
  const int lane  = threadIdx.x & 31;
  const float* row = x + (size_t)token * C_DIM;
  float v[6];
  float s = 0.f;
#pragma unroll
  for (int i = 0; i < 6; ++i) { v[i] = row[lane + 32 * i]; s += v[i]; }
#pragma unroll
  for (int o = 16; o > 0; o >>= 1) s += __shfl_xor(s, o, 32);
  const float mean = s * (1.f / 192.f);
  float var = 0.f;
#pragma unroll
  for (int i = 0; i < 6; ++i) { float d = v[i] - mean; var += d * d; }
#pragma unroll
  for (int o = 16; o > 0; o >>= 1) var += __shfl_xor(var, o, 32);
  const float rstd = rsqrtf(var * (1.f / 192.f) + 1e-5f);
#pragma unroll
  for (int i = 0; i < 6; ++i) {
    int c = lane + 32 * i;
    out[(size_t)token * C_DIM + c] = (_Float16)((v[i] - mean) * rstd * g[c] + b[c]);
  }
}

// ---------------- WMMA GEMM: Y[M][Nc] = X[M][Kc] @ Wt[Nc][Kc]^T + bias (opt gelu/res) ----
// Block = 256 threads (8 waves). Tile: 256(M) x 64(N) x 32(K), double-buffered async LDS.
// Wave w owns rows [w*32, w*32+32) (2 M-subtiles), all 64 columns -> 8 accumulators.
template <bool GELU, bool RES, bool OUTF32>
__global__ __launch_bounds__(256) void gemm_wmma(const _Float16* __restrict__ X,
                                                 const _Float16* __restrict__ Wt,
                                                 const float* __restrict__ bias,
                                                 const float* __restrict__ res,
                                                 _Float16* __restrict__ out16,
                                                 float* __restrict__ out32,
                                                 int Kc, int Nc) {
  __shared__ _Float16 As[2][256 * 32];   // 2 x 16 KB
  __shared__ _Float16 Bs[2][64 * 32];    // 2 x 4 KB

  const int m0   = blockIdx.x * 256;
  const int n0   = blockIdx.y * 64;
  const int tid  = threadIdx.x;
  const int wave = tid >> 5;
  const int lane = tid & 31;

  v8f acc[2][4];
#pragma unroll
  for (int i = 0; i < 2; ++i)
#pragma unroll
    for (int j = 0; j < 4; ++j)
#pragma unroll
      for (int r = 0; r < 8; ++r) acc[i][j][r] = 0.f;

  // stage one 256x32 A tile + 64x32 B tile into buffer `b` (async where available)
  auto stage = [&](int k0, int b) {
#pragma unroll
    for (int it = 0; it < 4; ++it) {           // A: 1024 16B chunks / 256 threads
      int c  = tid + it * 256;
      int r  = c >> 2;
      int kk = (c & 3) * 8;
      cp16_lds(&As[b][r * 32 + kk], X + (size_t)(m0 + r) * Kc + k0 + kk);
    }
    {                                          // B: 256 16B chunks / 256 threads
      int r  = tid >> 2;
      int kk = (tid & 3) * 8;
      cp16_lds(&Bs[b][r * 32 + kk], Wt + (size_t)(n0 + r) * Kc + k0 + kk);
    }
  };

  stage(0, 0);
  async_tile_wait();
  __syncthreads();

  int buf = 0;
  for (int k0 = 0; k0 < Kc; k0 += 32) {
    if (k0 + 32 < Kc) {
      stage(k0 + 32, buf ^ 1);                 // overlap copy of next tile with WMMAs
      __builtin_prefetch(X + (size_t)(m0 + (tid >> 1)) * Kc + k0 + 64, 0, 1);
    }

    v16h bfr[4];
#pragma unroll
    for (int j = 0; j < 4; ++j)
      bfr[j] = load_frag(&Bs[buf][(j * 16) * 32], 32, lane);
#pragma unroll
    for (int i = 0; i < 2; ++i) {
      v16h a = load_frag(&As[buf][((wave * 2 + i) * 16) * 32], 32, lane);
#pragma unroll
      for (int j = 0; j < 4; ++j)
        acc[i][j] = __builtin_amdgcn_wmma_f32_16x16x32_f16(false, a, false, bfr[j],
                                                           (short)0, acc[i][j], false, false);
    }

    async_tile_wait();
    __syncthreads();
    buf ^= 1;
  }

  // epilogue: D layout = VGPR r -> rows (r, r+8), col = lane&15
  const int cn = lane & 15;
  const int mh = (lane >> 4) * 8;
#pragma unroll
  for (int i = 0; i < 2; ++i) {
#pragma unroll
    for (int j = 0; j < 4; ++j) {
      int col = n0 + j * 16 + cn;
      float bv = bias[col];
#pragma unroll
      for (int r = 0; r < 8; ++r) {
        int row = m0 + (wave * 2 + i) * 16 + mh + r;
        float v = acc[i][j][r] + bv;
        if (GELU) v = 0.5f * v * (1.f + erff(v * 0.70710678118654752f));
        size_t idx = (size_t)row * Nc + col;
        if (RES) v += res[idx];
        if (OUTF32) out32[idx] = v;
        else        out16[idx] = (_Float16)v;
      }
    }
  }
}

// ---------------- attention: 1 block per window, 6 waves = 6 heads -----------------------
__global__ __launch_bounds__(192) void attn_kernel(const _Float16* __restrict__ qkv,
                                                   const float* __restrict__ mask,
                                                   const int* __restrict__ rel_pos,
                                                   const float* __restrict__ rpe_table,
                                                   _Float16* __restrict__ out) {
  __shared__ float    tbl[3 * RPEN * H_N];          // 3672 B
  __shared__ _Float16 Qs[H_N][KW * HDIM];           // 12 KB
  __shared__ _Float16 Ks[H_N][KW * HDIM];           // 12 KB
  __shared__ _Float16 Vt[H_N][KW * HDIM];           // 12 KB (transposed: [d][k])
  __shared__ _Float16 Ps[H_N][KW * KW];             // 12 KB
  __shared__ float    Ss[H_N][KW * 33];             // 25 KB

  const int win  = blockIdx.x;
  const int tid  = threadIdx.x;
  const int h    = tid >> 5;
  const int lane = tid & 31;

  for (int i = tid; i < 3 * RPEN * H_N; i += 192) tbl[i] = rpe_table[i];

  // stage Q (scaled), K, V^T for this head
  const size_t base = (size_t)win * KW * (3 * C_DIM);
  for (int i = lane; i < KW * HDIM; i += 32) {
    int t = i >> 5, d = i & 31;
    size_t row = base + (size_t)t * (3 * C_DIM) + h * HDIM + d;
    Qs[h][t * HDIM + d] = (_Float16)((float)qkv[row] * ATT_SCALE);
    Ks[h][t * HDIM + d] = qkv[row + C_DIM];
    Vt[h][d * KW + t]   = qkv[row + 2 * C_DIM];
  }
  __syncthreads();

  const int cn = lane & 15;
  const int mh = (lane >> 4) * 8;

  // S = Q @ K^T : 2x2 tiles of 16x16, K-dim = head dim = 32 -> one WMMA each
#pragma unroll
  for (int m0 = 0; m0 < 2; ++m0) {
    v16h a = load_frag(&Qs[h][m0 * 16 * HDIM], HDIM, lane);
#pragma unroll
    for (int nb = 0; nb < 2; ++nb) {
      v16h b = load_frag(&Ks[h][nb * 16 * HDIM], HDIM, lane);
      v8f acc;
#pragma unroll
      for (int r = 0; r < 8; ++r) acc[r] = 0.f;
      acc = __builtin_amdgcn_wmma_f32_16x16x32_f16(false, a, false, b,
                                                   (short)0, acc, false, false);
#pragma unroll
      for (int r = 0; r < 8; ++r)
        Ss[h][(m0 * 16 + mh + r) * 33 + nb * 16 + cn] = acc[r];
    }
  }
  __syncthreads();

  // softmax row per lane (lane == query row) with RPE gather + mask
  {
    const int q = lane;
    const size_t mbase = ((size_t)win * KW + q) * KW;
    const size_t rbase = mbase * 3;
    float s[KW];
    float mx = -1e30f;
#pragma unroll 4
    for (int k = 0; k < KW; ++k) {
      float v = Ss[h][q * 33 + k] + mask[mbase + k];
#pragma unroll
      for (int a = 0; a < 3; ++a) {
        int p = rel_pos[rbase + (size_t)k * 3 + a];
        p = (p < -POSB) ? -POSB : (p > POSB ? POSB : p);
        v += tbl[(p + POSB + a * RPEN) * H_N + h];
      }
      s[k] = v;
      mx = fmaxf(mx, v);
    }
    float sum = 0.f;
#pragma unroll 4
    for (int k = 0; k < KW; ++k) { s[k] = __expf(s[k] - mx); sum += s[k]; }
    const float inv = 1.f / sum;
#pragma unroll 4
    for (int k = 0; k < KW; ++k) Ps[h][q * KW + k] = (_Float16)(s[k] * inv);
  }
  __syncthreads();

  // O = P @ V : A = P (q x key), B = V^T rows are d -> contiguous key fragments
#pragma unroll
  for (int m0 = 0; m0 < 2; ++m0) {
    v16h a = load_frag(&Ps[h][m0 * 16 * KW], KW, lane);
#pragma unroll
    for (int nb = 0; nb < 2; ++nb) {
      v16h b = load_frag(&Vt[h][nb * 16 * KW], KW, lane);
      v8f acc;
#pragma unroll
      for (int r = 0; r < 8; ++r) acc[r] = 0.f;
      acc = __builtin_amdgcn_wmma_f32_16x16x32_f16(false, a, false, b,
                                                   (short)0, acc, false, false);
#pragma unroll
      for (int r = 0; r < 8; ++r) {
        int q = m0 * 16 + mh + r;
        int d = nb * 16 + cn;
        out[((size_t)win * KW + q) * C_DIM + h * HDIM + d] = (_Float16)acc[r];
      }
    }
  }
}

// ---------------- launch ----------------
extern "C" void kernel_launch(void* const* d_in, const int* in_sizes, int n_in,
                              void* d_out, int out_size, void* d_ws, size_t ws_size,
                              hipStream_t stream) {
  (void)in_sizes; (void)n_in; (void)out_size; (void)ws_size;
  const float* data  = (const float*)d_in[0];
  const float* mask  = (const float*)d_in[1];
  const int*   relp  = (const int*)d_in[2];
  const float* ln1g  = (const float*)d_in[3];
  const float* ln1b  = (const float*)d_in[4];
  const float* qkvw  = (const float*)d_in[5];
  const float* qkvb  = (const float*)d_in[6];
  const float* rpet  = (const float*)d_in[7];
  const float* projw = (const float*)d_in[8];
  const float* projb = (const float*)d_in[9];
  const float* ln2g  = (const float*)d_in[10];
  const float* ln2b  = (const float*)d_in[11];
  const float* w1    = (const float*)d_in[12];
  const float* b1    = (const float*)d_in[13];
  const float* w2    = (const float*)d_in[14];
  const float* b2    = (const float*)d_in[15];
  float* out = (float*)d_out;

  // workspace layout
  char* ws = (char*)d_ws;
  _Float16* qkv_wt  = (_Float16*)ws;                         // [576][192]
  _Float16* proj_wt = qkv_wt + 576 * 192;                    // [192][192]
  _Float16* w1t     = proj_wt + 192 * 192;                   // [768][192]
  _Float16* w2t     = w1t + 192 * 768;                       // [192][768]
  size_t off = (size_t)1 << 20;
  _Float16* R1 = (_Float16*)(ws + off);                      // qkv f16 / mlp hidden f16
  _Float16* R2 = (_Float16*)(ws + off + (size_t)MTOT * HID * 2);   // ln-out / attn-out f16
  float*    R3 = (float*)(ws + off + (size_t)MTOT * HID * 2 + (size_t)MTOT * C_DIM * 2); // res1 f32

  // 1. weight conversion to f16 transposed
  convert_w<<<(192 * 576 + 255) / 256, 256, 0, stream>>>(qkvw, qkv_wt, 192, 576);
  convert_w<<<(192 * 192 + 255) / 256, 256, 0, stream>>>(projw, proj_wt, 192, 192);
  convert_w<<<(192 * 768 + 255) / 256, 256, 0, stream>>>(w1, w1t, 192, 768);
  convert_w<<<(768 * 192 + 255) / 256, 256, 0, stream>>>(w2, w2t, 768, 192);

  // 2. LN1 -> R2
  ln_kernel<<<MTOT / 8, 256, 0, stream>>>(data, ln1g, ln1b, R2);
  // 3. QKV GEMM -> R1
  {
    dim3 g(MTOT / 256, 576 / 64);
    gemm_wmma<false, false, false><<<g, 256, 0, stream>>>(R2, qkv_wt, qkvb, nullptr, R1, nullptr, 192, 576);
  }
  // 4. attention -> R2
  attn_kernel<<<N_WIN, 192, 0, stream>>>(R1, mask, relp, rpet, R2);
  // 5. proj + residual(data) -> R3 (f32)
  {
    dim3 g(MTOT / 256, 192 / 64);
    gemm_wmma<false, true, true><<<g, 256, 0, stream>>>(R2, proj_wt, projb, data, nullptr, R3, 192, 192);
  }
  // 6. LN2 -> R2
  ln_kernel<<<MTOT / 8, 256, 0, stream>>>(R3, ln2g, ln2b, R2);
  // 7. MLP1 + GELU -> R1
  {
    dim3 g(MTOT / 256, 768 / 64);
    gemm_wmma<true, false, false><<<g, 256, 0, stream>>>(R2, w1t, b1, nullptr, R1, nullptr, 192, 768);
  }
  // 8. MLP2 + residual(R3) -> d_out (f32)
  {
    dim3 g(MTOT / 256, 192 / 64);
    gemm_wmma<false, true, true><<<g, 256, 0, stream>>>(R1, w2t, b2, R3, nullptr, out, 768, 192);
  }
}
